// WindowMSA_82824149336842
// MI455X (gfx1250) — matrix operand order repaired and checked by reading, not compiled, not run on hardware
//
#include <hip/hip_runtime.h>

// ---------------- types ----------------
typedef __bf16 bf16_t;
typedef bf16_t bf16x16 __attribute__((ext_vector_type(16)));
typedef bf16_t bf16x8  __attribute__((ext_vector_type(8)));
typedef bf16_t bf16x4  __attribute__((ext_vector_type(4)));
typedef float  f32x8   __attribute__((ext_vector_type(8)));

#define WMMA_BF16(a, b, c) \
  __builtin_amdgcn_wmma_f32_16x16x32_bf16(false, (a), false, (b), (short)0, (c), false, false)

// native hardware convert (RTNE)
__device__ __forceinline__ bf16_t f2bf(float f) { return (bf16_t)f; }

// A fragment: 16x32 bf16, row-major MxK source with leading dim `ld` (elements).
// lane = g*16 + m ; elements e=0..7 -> K=g*8+e ; e=8..15 -> K=g*8+16+(e-8)
__device__ __forceinline__ bf16x16 frag_a(const bf16_t* p, int ld) {
  const int lane = threadIdx.x & 31;
  const int m = lane & 15, g = lane >> 4;
  const bf16_t* row = p + m * ld + g * 8;
  bf16x8 lo = *(const bf16x8*)(row);
  bf16x8 hi = *(const bf16x8*)(row + 16);
  bf16x16 r;
#pragma unroll
  for (int i = 0; i < 8; ++i) { r[i] = lo[i]; r[i + 8] = hi[i]; }
  return r;
}

// B fragment: 32x16 bf16 (KxN), source stored N-major (N rows of K) with ld elems.
// lane = g*16 + n ; elements e=0..7 -> K=g*8+e ; e=8..15 -> K=g*8+16+(e-8)
__device__ __forceinline__ bf16x16 frag_b(const bf16_t* p, int ld) {
  const int lane = threadIdx.x & 31;
  const int n = lane & 15, g = lane >> 4;
  const bf16_t* row = p + n * ld + g * 8;
  bf16x8 lo = *(const bf16x8*)(row);
  bf16x8 hi = *(const bf16x8*)(row + 16);
  bf16x16 r;
#pragma unroll
  for (int i = 0; i < 8; ++i) { r[i] = lo[i]; r[i + 8] = hi[i]; }
  return r;
}

// ---------------- static config ----------------
// B0=32, H=W=64, C=128, WS=8, HEADS=4, SHIFT=4, T=64, NW=64, hd=32

// workspace layout (bytes)
#define WS_QKVT   0         // bf16 [384][128]  = 98304 B   (qkv_w transposed, N-major)
#define WS_PROJT  98304     // bf16 [128][128]  = 32768 B   (proj_w transposed, N-major)
#define WS_RPB    131072    // f32  [4][64][64] = 65536 B   (gathered rel-pos bias)

// LDS layout (bytes), aliased by liveness
#define SM_VT    0        // bf16 [4][32][64]  = 16384   (V transposed: head,hd,s)  S2..S5
#define SM_Q     16384    // bf16 [4][64][32]  = 16384                              S2..S3
#define SM_K     32768    // bf16 [4][64][32]  = 16384                              S2..S3
#define SM_XW    49152    // bf16 [64][136]    = 17408                              S1..S2
#define SM_LOG   49152    // f32  [4][64][64]  = 65536   (aliases XW)               S3..S4
#define SM_PROB  16384    // bf16 [4][64][64]  = 32768   (aliases Q,K)              S4..S5
#define SM_CTX   49152    // bf16 [64][136]    = 17408   (aliases LOG)              S5..S6
#define SMEM_BYTES 114688

// ---------------- prep: weight transpose->bf16, rel-pos-bias gather ----------------
__global__ __launch_bounds__(256) void swin_prep(const float* __restrict__ qkv_w,
                                                 const float* __restrict__ proj_w,
                                                 const float* __restrict__ bias_table,
                                                 const int* __restrict__ rel_index,
                                                 char* __restrict__ ws) {
  bf16_t* qkvT  = (bf16_t*)(ws + WS_QKVT);
  bf16_t* projT = (bf16_t*)(ws + WS_PROJT);
  float*  rpb   = (float*)(ws + WS_RPB);
  int i = blockIdx.x * 256 + threadIdx.x;
  if (i < 49152) {                       // qkv_wT[n][k] = qkv_w[k][n], n<384,k<128
    int n = i >> 7, k = i & 127;
    qkvT[i] = f2bf(qkv_w[k * 384 + n]);
  } else if (i < 65536) {                // proj_wT[n][k] = proj_w[k][n]
    int j = i - 49152;
    int n = j >> 7, k = j & 127;
    projT[j] = f2bf(proj_w[k * 128 + n]);
  } else if (i < 81920) {                // rpb[h][t][s] = bias_table[rel_index[t][s]][h]
    int j = i - 65536;
    int h = j >> 12, ts = j & 4095;
    rpb[j] = bias_table[rel_index[ts] * 4 + h];
  }
}

// ---------------- fused shifted-window MSA: one block per window ----------------
__global__ __launch_bounds__(256) void swin_msa(const float* __restrict__ x,
                                                const float* __restrict__ qkv_b,
                                                const float* __restrict__ proj_b,
                                                const float* __restrict__ attn_mask,
                                                const char* __restrict__ ws,
                                                float* __restrict__ out) {
  extern __shared__ char smem[];
  bf16_t* vT = (bf16_t*)(smem + SM_VT);
  bf16_t* qS = (bf16_t*)(smem + SM_Q);
  bf16_t* kS = (bf16_t*)(smem + SM_K);
  bf16_t* xw = (bf16_t*)(smem + SM_XW);
  float*  lg = (float*)(smem + SM_LOG);
  bf16_t* pr = (bf16_t*)(smem + SM_PROB);
  bf16_t* cx = (bf16_t*)(smem + SM_CTX);

  const bf16_t* qkvT  = (const bf16_t*)(ws + WS_QKVT);
  const bf16_t* projT = (const bf16_t*)(ws + WS_PROJT);
  const float*  rpb   = (const float*)(ws + WS_RPB);

  const int tid   = threadIdx.x;
  const int lane  = tid & 31;
  const int nn    = lane & 15;          // column within tile
  const int halfm = lane >> 4;          // row-half selector
  const int wave  = __builtin_amdgcn_readfirstlane(tid >> 5);  // wave-uniform -> SGPR
  const int bw = blockIdx.x;
  const int b  = bw >> 6;               // image
  const int wi = bw & 63;               // window within image
  const int wy = wi >> 3, wx = wi & 7;  // window grid coords

  // warm L2/WGP$ for the shared bf16 weights (global_prefetch_b8)
  __builtin_prefetch(qkvT + (tid << 6), 0, 0);
  __builtin_prefetch(projT + (tid << 5), 0, 0);

  // ---- S1: gather shifted window (roll -4,-4), f32 -> bf16 into LDS ----
  const float4* x4 = (const float4*)x;
#pragma unroll
  for (int i = 0; i < 8; ++i) {
    int e  = tid + i * 256;      // 2048 float4 slots = 64 tok * 32
    int tt = e >> 5;             // token
    int c4 = e & 31;             // float4 index within 128 channels
    int ty = tt >> 3, tx = tt & 7;
    int h = (wy * 8 + ty + 4) & 63;   // roll(-SHIFT): src = (dst+SHIFT) mod H
    int w = (wx * 8 + tx + 4) & 63;
    float4 f = x4[((size_t)b * 4096 + h * 64 + w) * 32 + c4];
    bf16x4 o; o[0] = f2bf(f.x); o[1] = f2bf(f.y); o[2] = f2bf(f.z); o[3] = f2bf(f.w);
    *(bf16x4*)(xw + tt * 136 + c4 * 4) = o;
  }
  __syncthreads();

  // ---- S2: QKV = xw(64x128) @ Wqkv(128x384) + b ; emit q (scaled), k, v^T ----
  const float inv_sqrt_hd = 0.17677669529663687f;   // 32^-0.5
#pragma unroll
  for (int mt = 0; mt < 4; ++mt) {
    bf16x16 a0 = frag_a(xw + mt * 16 * 136 + 0,  136);
    bf16x16 a1 = frag_a(xw + mt * 16 * 136 + 32, 136);
    bf16x16 a2 = frag_a(xw + mt * 16 * 136 + 64, 136);
    bf16x16 a3 = frag_a(xw + mt * 16 * 136 + 96, 136);
#pragma unroll
    for (int t3 = 0; t3 < 3; ++t3) {
      const int nt = __builtin_amdgcn_readfirstlane(wave * 3 + t3); // tile col, scalar
      const bf16_t* wp = qkvT + nt * 16 * 128;
      f32x8 acc = {};
      acc = WMMA_BF16(a0, frag_b(wp + 0,  128), acc);
      acc = WMMA_BF16(a1, frag_b(wp + 32, 128), acc);
      acc = WMMA_BF16(a2, frag_b(wp + 64, 128), acc);
      acc = WMMA_BF16(a3, frag_b(wp + 96, 128), acc);
      // tile-uniform section/head (16-col tile never crosses a 32-col boundary)
      const int colbase = nt * 16;
      const int sec    = colbase >> 7;          // 0=q 1=k 2=v  (scalar)
      const int head   = (colbase >> 5) & 3;    // scalar
      const int hibase = colbase & 31;          // 0 or 16      (scalar)
      const int hi     = hibase + nn;
      const int tokbase = mt * 16 + halfm * 8;
      float bias = qkv_b[colbase + nn];
      if (sec == 0) {
#pragma unroll
        for (int r = 0; r < 8; ++r)
          qS[head * 2048 + (tokbase + r) * 32 + hi] = f2bf((acc[r] + bias) * inv_sqrt_hd);
      } else if (sec == 1) {
#pragma unroll
        for (int r = 0; r < 8; ++r)
          kS[head * 2048 + (tokbase + r) * 32 + hi] = f2bf(acc[r] + bias);
      } else {
        bf16x8 o;
#pragma unroll
        for (int r = 0; r < 8; ++r) o[r] = f2bf(acc[r] + bias);
        *(bf16x8*)(vT + head * 2048 + hi * 64 + tokbase) = o;   // 8 consecutive tokens
      }
    }
  }
  __syncthreads();

  // ---- S3: logits = q @ k^T + rpb + mask  (K = hd = 32 -> single WMMA/tile) ----
  {
    const int head = __builtin_amdgcn_readfirstlane(wave >> 1);
    const int base = __builtin_amdgcn_readfirstlane((wave & 1) * 8);
    const float* rpbh  = rpb + head * 4096;
    const float* maskw = attn_mask + wi * 4096;
    float* lgh = lg + head * 4096;
#pragma unroll
    for (int j = 0; j < 8; ++j) {
      int idx = base + j;
      int mt = idx >> 2, st = idx & 3;
      bf16x16 a  = frag_a(qS + head * 2048 + mt * 16 * 32, 32);
      bf16x16 bb = frag_b(kS + head * 2048 + st * 16 * 32, 32);
      f32x8 acc = {};
      acc = WMMA_BF16(a, bb, acc);
      int s = st * 16 + nn;
      int tbase = mt * 16 + halfm * 8;
#pragma unroll
      for (int r = 0; r < 8; ++r) {
        int off = (tbase + r) * 64 + s;
        lgh[off] = acc[r] + rpbh[off] + maskw[off];
      }
    }
  }
  __syncthreads();

  // ---- S4: softmax over s, one thread per (head, t) row; write bf16 probs ----
  {
    int head = tid >> 6, t = tid & 63;
    float* row = lg + head * 4096 + t * 64;
    float m = -1e30f;
    for (int s = 0; s < 64; ++s) m = fmaxf(m, row[s]);
    float sum = 0.f;
    for (int s = 0; s < 64; ++s) { float e = __expf(row[s] - m); row[s] = e; sum += e; }
    float inv = 1.0f / sum;
    bf16_t* prow = pr + head * 4096 + t * 64;
    for (int s = 0; s < 64; ++s) prow[s] = f2bf(row[s] * inv);
  }
  __syncthreads();

  // ---- S5: ctx = probs(64x64) @ v(64x32) per head ----
  {
    const int head = __builtin_amdgcn_readfirstlane(wave >> 1);
    const int nt2  = __builtin_amdgcn_readfirstlane(wave & 1);  // hd half
#pragma unroll
    for (int mt = 0; mt < 4; ++mt) {
      f32x8 acc = {};
#pragma unroll
      for (int ks = 0; ks < 2; ++ks) {
        bf16x16 a  = frag_a(pr + head * 4096 + mt * 16 * 64 + ks * 32, 64);
        bf16x16 bb = frag_b(vT + head * 2048 + nt2 * 16 * 64 + ks * 32, 64);
        acc = WMMA_BF16(a, bb, acc);
      }
      int col = head * 32 + nt2 * 16 + nn;
      int tokbase = mt * 16 + halfm * 8;
#pragma unroll
      for (int r = 0; r < 8; ++r)
        cx[(tokbase + r) * 136 + col] = f2bf(acc[r]);
    }
  }
  __syncthreads();

  // ---- S6: out = ctx(64x128) @ proj(128x128) + b ; window-reverse + roll(+4,+4) ----
  {
    const int nt = wave;                      // 8 N tiles of 16 over 128 cols (scalar)
    const bf16_t* wp = projT + nt * 16 * 128;
    const int wbase = wx * 8 + 4;
#pragma unroll
    for (int mt = 0; mt < 4; ++mt) {
      f32x8 acc = {};
      acc = WMMA_BF16(frag_a(cx + mt * 16 * 136 + 0,  136), frag_b(wp + 0,  128), acc);
      acc = WMMA_BF16(frag_a(cx + mt * 16 * 136 + 32, 136), frag_b(wp + 32, 128), acc);
      acc = WMMA_BF16(frag_a(cx + mt * 16 * 136 + 64, 136), frag_b(wp + 64, 128), acc);
      acc = WMMA_BF16(frag_a(cx + mt * 16 * 136 + 96, 136), frag_b(wp + 96, 128), acc);
      int col = nt * 16 + nn;
      float bias = proj_b[col];
      int tokbase = mt * 16 + halfm * 8;      // multiple of 8 -> ty invariant, tx = r
      int ty = tokbase >> 3;
      int h = (wy * 8 + ty + 4) & 63;         // reverse roll(+SHIFT)
      float* orow = out + ((size_t)b * 4096 + h * 64) * 128 + col;
#pragma unroll
      for (int r = 0; r < 8; ++r) {
        int w = (wbase + r) & 63;
        orow[w * 128] = acc[r] + bias;
      }
    }
  }
}

// ---------------- host launcher ----------------
extern "C" void kernel_launch(void* const* d_in, const int* in_sizes, int n_in,
                              void* d_out, int out_size, void* d_ws, size_t ws_size,
                              hipStream_t stream) {
  const float* x          = (const float*)d_in[0];
  const float* qkv_w      = (const float*)d_in[1];
  const float* qkv_b      = (const float*)d_in[2];
  const float* proj_w     = (const float*)d_in[3];
  const float* proj_b     = (const float*)d_in[4];
  const float* bias_table = (const float*)d_in[5];
  const int*   rel_index  = (const int*)d_in[6];
  const float* attn_mask  = (const float*)d_in[7];
  float* out = (float*)d_out;
  char*  ws  = (char*)d_ws;
  (void)in_sizes; (void)n_in; (void)out_size; (void)ws_size;

  swin_prep<<<320, 256, 0, stream>>>(qkv_w, proj_w, bias_table, rel_index, ws);
  swin_msa<<<2048, 256, SMEM_BYTES, stream>>>(x, qkv_b, proj_b, attn_mask, ws, out);
}
